// EnhancedGAT_18193481466335
// MI455X (gfx1250) — matrix operand order repaired and checked by reading, not compiled, not run on hardware
//
#include <hip/hip_runtime.h>
#include <hip/hip_bf16.h>
#include <math.h>

// Problem constants (fixed by the reference).
#define KIN    128
#define HIDC   64
#define NHEADS 4
#define C1     256        // NHEADS*HIDC
#define NCLS   40
#define NT2    3          // n-tiles for the 40(->48)-wide GEMMs
#define SLOPE  0.2f
#define BNEPS  1e-5f

typedef __bf16 bf16_t;
typedef __attribute__((ext_vector_type(16))) __bf16 v16bf;
typedef __attribute__((ext_vector_type(8)))  __bf16 v8bf;
typedef __attribute__((ext_vector_type(8)))  float  v8f;

// ---------------------------------------------------------------- utilities
__global__ void fill_f32(float* __restrict__ p, float v, long long n) {
  long long i = (long long)blockIdx.x * blockDim.x + threadIdx.x;
  if (i < n) p[i] = v;
}

__global__ void cvt_bf16(bf16_t* __restrict__ dst, const float* __restrict__ src,
                         long long n) {
  long long i = (long long)blockIdx.x * blockDim.x + threadIdx.x;
  if (i < n) dst[i] = (bf16_t)src[i];
}

// Re-pack f32 weight [K, cols] row-major into the WMMA B-fragment layout:
// dst[(((nt*KB + kb)*32 + lane)*16 + e)] = B[kb*32 + (lane>>4)*8 + (e<8 ? e : e+8), nt*16 + (lane&15)]
// so each lane's 16-element fragment is one contiguous 32-byte load.
__global__ void pack_b_frag(bf16_t* __restrict__ dst, const float* __restrict__ src,
                            int K, int cols, int ntiles) {
  int KB = K >> 5;
  long long total = (long long)ntiles * KB * 512;
  long long t = (long long)blockIdx.x * blockDim.x + threadIdx.x;
  if (t >= total) return;
  int e    = (int)(t & 15);
  int lane = (int)((t >> 4) & 31);
  long long tmp = t >> 9;
  int kb = (int)(tmp % KB);
  int nt = (int)(tmp / KB);
  int k   = kb * 32 + (lane >> 4) * 8 + (e < 8 ? e : e + 8);
  int col = nt * 16 + (lane & 15);
  dst[t] = (col < cols) ? (bf16_t)src[(size_t)k * cols + col] : (bf16_t)0.0f;
}

// ------------------------------------------------------------- WMMA GEMM
// C[M,Nc](ldc) = A[M,K] (bf16 row-major) x Bpk (fragment-packed, ntiles*K/32*512).
// One wave per 16x16 tile; K multiple of 32.
__global__ __launch_bounds__(256)
void gemm_bf16_wmma(const bf16_t* __restrict__ A, const bf16_t* __restrict__ Bpk,
                    float* __restrict__ C, int M, int K, int ntiles, int Nc, int ldc) {
  int wid    = blockIdx.x * (blockDim.x >> 5) + (threadIdx.x >> 5);
  int mtiles = (M + 15) >> 4;
  if (wid >= mtiles * ntiles) return;          // uniform per-wave exit (EXEC all-1 for WMMA)
  int mt = wid % mtiles, nt = wid / mtiles;
  int lane = threadIdx.x & 31;
  int r  = lane & 15;        // M (A) / N (B,C) index within tile
  int kh = lane >> 4;        // lane-half selector
  int KB = K >> 5;

  int arow = mt * 16 + r; if (arow >= M) arow = M - 1;   // clamp (discarded on store)
  // A 16-bit layout: lane=M; element e -> k = kh*8 + (e<8 ? e : e+8)  [two 8-chunks]
  const bf16_t* Ab = A + (size_t)arow * K + kh * 8;
  const bf16_t* Bb = Bpk + ((size_t)nt * KB * 32 + lane) * 16;

  v8f acc = {};
#pragma unroll 4
  for (int kb = 0; kb < KB; ++kb) {
    v8bf alo = *(const v8bf*)(Ab + kb * 32);        // k = kb*32 + kh*8 + 0..7
    v8bf ahi = *(const v8bf*)(Ab + kb * 32 + 16);   // k = kb*32 + kh*8 + 16..23
    v16bf a = __builtin_shufflevector(alo, ahi, 0, 1, 2, 3, 4, 5, 6, 7,
                                      8, 9, 10, 11, 12, 13, 14, 15);
    v16bf b = *(const v16bf*)(Bb + (size_t)kb * 512);
    if (kb + 1 < KB) {
      __builtin_prefetch(Ab + kb * 32 + 32, 0, 3);  // global_prefetch_b8
      __builtin_prefetch(Bb + (size_t)(kb + 1) * 512, 0, 3);
    }
    acc = __builtin_amdgcn_wmma_f32_16x16x32_bf16(
        /*neg_a=*/false, a, /*neg_b=*/false, b,
        /*c_mod=*/(short)0, acc, /*reuse_a=*/false, /*reuse_b=*/false);
  }

  // C/D layout: VGPR j -> M = j (lanes 0-15) or j+8 (lanes 16-31); N = lane&15.
  int col = nt * 16 + r;
  if (col < Nc) {
    int rbase = mt * 16 + (kh ? 8 : 0);
#pragma unroll
    for (int j = 0; j < 8; ++j) {
      int row = rbase + j;
      if (row < M) C[(size_t)row * ldc + col] = acc[j];
    }
  }
}

// -------------------------------------------------- attention coefficients
// out[n*heads+h] = sum_c h[n, h, c] * att[h, c]   (C multiple of 4)
__global__ void att_dot(const float* __restrict__ h, const float* __restrict__ att,
                        float* __restrict__ out, int n, int heads, int C) {
  int t = blockIdx.x * blockDim.x + threadIdx.x;
  if (t >= n * heads) return;
  int hh = t % heads, node = t / heads;
  const float4* hp = (const float4*)(h + (size_t)node * heads * C + hh * C);
  const float4* ap = (const float4*)(att + hh * C);
  float s = 0.f;
  for (int c = 0; c < (C >> 2); ++c) {
    float4 a = hp[c], b = ap[c];
    s += a.x * b.x + a.y * b.y + a.z * b.z + a.w * b.w;
  }
  out[t] = s;
}

// ------------------------------------------------------------ edge kernels
__device__ __forceinline__ void atomicMaxF(float* addr, float val) {
  // Bit trick: positive floats ordered as ints, negatives reversed as uints.
  if (val >= 0.f) atomicMax((int*)addr, __float_as_int(val));
  else            atomicMin((unsigned int*)addr, __float_as_uint(val));
}

__device__ __forceinline__ void edge_sd(const long long* ei, long long e,
                                        long long eorig, long long& s, long long& d) {
  if (e < eorig) { s = ei[e]; d = ei[eorig + e]; }
  else           { s = d = e - eorig; }          // self loops
}

__global__ void edge_max(const long long* __restrict__ ei,
                         const float* __restrict__ as, const float* __restrict__ ad,
                         float* __restrict__ m, int heads, long long etot, long long eorig) {
  long long t = (long long)blockIdx.x * blockDim.x + threadIdx.x;
  if (t >= etot * heads) return;
  long long e = t / heads; int hh = (int)(t % heads);
  long long s, d; edge_sd(ei, e, eorig, s, d);
  float v = as[s * heads + hh] + ad[d * heads + hh];
  v = v > 0.f ? v : SLOPE * v;
  atomicMaxF(&m[d * heads + hh], v);
}

__global__ void edge_expsum(const long long* __restrict__ ei,
                            const float* __restrict__ as, const float* __restrict__ ad,
                            const float* __restrict__ m, float* __restrict__ z,
                            int heads, long long etot, long long eorig) {
  long long t = (long long)blockIdx.x * blockDim.x + threadIdx.x;
  if (t >= etot * heads) return;
  long long e = t / heads; int hh = (int)(t % heads);
  long long s, d; edge_sd(ei, e, eorig, s, d);
  float v = as[s * heads + hh] + ad[d * heads + hh];
  v = v > 0.f ? v : SLOPE * v;
  atomicAdd(&z[d * heads + hh], expf(v - m[d * heads + hh]));
}

// One thread per (edge, 4 channels); heads*C multiple of 4, C multiple of 4.
__global__ void edge_aggr4(const long long* __restrict__ ei,
                           const float* __restrict__ as, const float* __restrict__ ad,
                           const float* __restrict__ m, const float* __restrict__ z,
                           const float* __restrict__ hsrc, float* __restrict__ out,
                           int heads, int C, long long etot, long long eorig) {
  int ct  = heads * C;
  int ctv = ct >> 2;
  long long t = (long long)blockIdx.x * blockDim.x + threadIdx.x;
  if (t >= etot * ctv) return;
  long long e = t / ctv;
  int c4 = ((int)(t % ctv)) << 2;
  int hh = c4 / C;
  long long s, d; edge_sd(ei, e, eorig, s, d);
  float v = as[s * heads + hh] + ad[d * heads + hh];
  v = v > 0.f ? v : SLOPE * v;
  float alpha = expf(v - m[d * heads + hh]) / z[d * heads + hh];
  float4 hv = *(const float4*)(hsrc + (size_t)s * ct + c4);
  float* op = out + (size_t)d * ct + c4;
  atomicAdd(op + 0, alpha * hv.x);
  atomicAdd(op + 1, alpha * hv.y);
  atomicAdd(op + 2, alpha * hv.z);
  atomicAdd(op + 3, alpha * hv.w);
}

// ------------------------------------------------------------- batch norm
__global__ __launch_bounds__(256)
void bn_stats(const float* __restrict__ x, float* __restrict__ mu,
              float* __restrict__ var, int n, int C) {
  int col = blockIdx.x;
  float s = 0.f, sq = 0.f;
  for (int r = threadIdx.x; r < n; r += blockDim.x) {
    float v = x[(size_t)r * C + col];
    s += v; sq += v * v;
  }
  __shared__ float ss[256], s2[256];
  ss[threadIdx.x] = s; s2[threadIdx.x] = sq;
  __syncthreads();
  for (int o = 128; o > 0; o >>= 1) {
    if ((int)threadIdx.x < o) { ss[threadIdx.x] += ss[threadIdx.x + o];
                                s2[threadIdx.x] += s2[threadIdx.x + o]; }
    __syncthreads();
  }
  if (threadIdx.x == 0) {
    float mm = ss[0] / n;
    mu[col]  = mm;
    var[col] = s2[0] / n - mm * mm;
  }
}

// normalize + ELU, emit bf16 for the next WMMA GEMM.
__global__ void bn_elu_cvt(const float* __restrict__ x, const float* __restrict__ mu,
                           const float* __restrict__ var, const float* __restrict__ gamma,
                           const float* __restrict__ beta, bf16_t* __restrict__ out,
                           long long n, int C) {
  long long t = (long long)blockIdx.x * blockDim.x + threadIdx.x;
  if (t >= n * C) return;
  int c = (int)(t % C);
  float v = x[t];
  v = (v - mu[c]) * rsqrtf(var[c] + BNEPS) * gamma[c] + beta[c];
  v = v > 0.f ? v : expf(v) - 1.f;               // ELU
  out[t] = (bf16_t)v;
}

// ---------------------------------------------------- skip + log_softmax
__global__ void final_logsoftmax(const float* __restrict__ gat2, const float* __restrict__ skip,
                                 const float* __restrict__ b2, const float* __restrict__ bskip,
                                 float* __restrict__ out, int n) {
  int node = blockIdx.x * blockDim.x + threadIdx.x;
  if (node >= n) return;
  float v[NCLS];
  float mx = -3.0e38f;
  for (int c = 0; c < NCLS; ++c) {
    float t = gat2[(size_t)node * NCLS + c] + b2[c] + skip[(size_t)node * NCLS + c] + bskip[c];
    v[c] = t; mx = fmaxf(mx, t);
  }
  float s = 0.f;
  for (int c = 0; c < NCLS; ++c) s += expf(v[c] - mx);
  float lg = logf(s);
  for (int c = 0; c < NCLS; ++c) out[(size_t)node * NCLS + c] = v[c] - mx - lg;
}

// ---------------------------------------------------------------- launcher
static inline unsigned gridFor(long long n, int tpb) {
  return (unsigned)((n + tpb - 1) / tpb);
}

extern "C" void kernel_launch(void* const* d_in, const int* in_sizes, int n_in,
                              void* d_out, int out_size, void* d_ws, size_t ws_size,
                              hipStream_t stream) {
  const float*     x       = (const float*)d_in[0];
  const long long* ei      = (const long long*)d_in[1];   // int64 [2, E]
  const float*     W1      = (const float*)d_in[2];
  const float*     attS1   = (const float*)d_in[3];
  const float*     attD1   = (const float*)d_in[4];
  /* b1 (d_in[5]) is mathematically cancelled by batch-norm mean subtraction */
  const float*     gamma   = (const float*)d_in[6];
  const float*     beta    = (const float*)d_in[7];
  const float*     W2      = (const float*)d_in[8];
  const float*     attS2   = (const float*)d_in[9];
  const float*     attD2   = (const float*)d_in[10];
  const float*     b2      = (const float*)d_in[11];
  const float*     Wskip   = (const float*)d_in[12];
  const float*     bskip   = (const float*)d_in[13];
  float*           out     = (float*)d_out;

  const int       N    = in_sizes[0] / KIN;      // 50000
  const long long E    = in_sizes[1] / 2;        // 800000
  const long long ETOT = E + N;                  // with self loops
  const int       NT1  = C1 / 16;                // 16 n-tiles for GEMM1

  // ---- workspace carve-out (deterministic, 256B aligned) ----
  char* ws = (char*)d_ws;
  size_t off = 0;
  auto carve = [&](size_t bytes) -> void* {
    void* p = ws + off;
    off = (off + bytes + 255) & ~(size_t)255;
    return p;
  };
  bf16_t* xb      = (bf16_t*)carve((size_t)N * KIN * 2);
  bf16_t* W1pk    = (bf16_t*)carve((size_t)NT1 * (KIN / 32) * 512 * 2);
  float*  h1      = (float*) carve((size_t)N * C1 * 4);
  float*  as1     = (float*) carve((size_t)N * NHEADS * 4);
  float*  ad1     = (float*) carve((size_t)N * NHEADS * 4);
  float*  m1      = (float*) carve((size_t)N * NHEADS * 4);
  float*  z1      = (float*) carve((size_t)N * NHEADS * 4);
  float*  out1    = (float*) carve((size_t)N * C1 * 4);
  float*  mu      = (float*) carve((size_t)C1 * 4);
  float*  var     = (float*) carve((size_t)C1 * 4);
  bf16_t* x1b     = (bf16_t*)carve((size_t)N * C1 * 2);
  bf16_t* W2pk    = (bf16_t*)carve((size_t)NT2 * (C1 / 32) * 512 * 2);
  float*  h2      = (float*) carve((size_t)N * NCLS * 4);
  float*  as2     = (float*) carve((size_t)N * 4);
  float*  ad2     = (float*) carve((size_t)N * 4);
  float*  m2      = (float*) carve((size_t)N * 4);
  float*  z2      = (float*) carve((size_t)N * 4);
  float*  out2    = (float*) carve((size_t)N * NCLS * 4);
  bf16_t* Wskippk = (bf16_t*)carve((size_t)NT2 * (KIN / 32) * 512 * 2);
  float*  hskip   = (float*) carve((size_t)N * NCLS * 4);

  const int TPB = 256;

  // ---- stage 0: bf16 A-copy of x, fragment-pack all weights ----
  cvt_bf16<<<gridFor((long long)N * KIN, TPB), TPB, 0, stream>>>(xb, x, (long long)N * KIN);
  pack_b_frag<<<gridFor((long long)NT1 * (KIN / 32) * 512, TPB), TPB, 0, stream>>>(
      W1pk, W1, KIN, C1, NT1);
  pack_b_frag<<<gridFor((long long)NT2 * (C1 / 32) * 512, TPB), TPB, 0, stream>>>(
      W2pk, W2, C1, NCLS, NT2);
  pack_b_frag<<<gridFor((long long)NT2 * (KIN / 32) * 512, TPB), TPB, 0, stream>>>(
      Wskippk, Wskip, KIN, NCLS, NT2);

  // ---- GEMM1: h1 = x @ W1  (50000x128x256) ----
  {
    long long tiles = ((N + 15) / 16) * (long long)NT1;
    gemm_bf16_wmma<<<gridFor(tiles, 8), TPB, 0, stream>>>(xb, W1pk, h1, N, KIN, NT1, C1, C1);
  }

  // ---- layer-1 attention coefficients ----
  att_dot<<<gridFor((long long)N * NHEADS, TPB), TPB, 0, stream>>>(h1, attS1, as1, N, NHEADS, HIDC);
  att_dot<<<gridFor((long long)N * NHEADS, TPB), TPB, 0, stream>>>(h1, attD1, ad1, N, NHEADS, HIDC);

  // ---- layer-1 segment softmax + aggregation ----
  fill_f32<<<gridFor((long long)N * NHEADS, TPB), TPB, 0, stream>>>(m1, -INFINITY, (long long)N * NHEADS);
  fill_f32<<<gridFor((long long)N * NHEADS, TPB), TPB, 0, stream>>>(z1, 0.f, (long long)N * NHEADS);
  fill_f32<<<gridFor((long long)N * C1, TPB), TPB, 0, stream>>>(out1, 0.f, (long long)N * C1);
  edge_max   <<<gridFor(ETOT * NHEADS, TPB), TPB, 0, stream>>>(ei, as1, ad1, m1, NHEADS, ETOT, E);
  edge_expsum<<<gridFor(ETOT * NHEADS, TPB), TPB, 0, stream>>>(ei, as1, ad1, m1, z1, NHEADS, ETOT, E);
  edge_aggr4 <<<gridFor(ETOT * (C1 / 4), TPB), TPB, 0, stream>>>(ei, as1, ad1, m1, z1, h1, out1,
                                                                 NHEADS, HIDC, ETOT, E);

  // ---- batch norm + ELU (+ bf16 for next GEMM) ----
  bn_stats<<<C1, TPB, 0, stream>>>(out1, mu, var, N, C1);
  bn_elu_cvt<<<gridFor((long long)N * C1, TPB), TPB, 0, stream>>>(out1, mu, var, gamma, beta,
                                                                  x1b, (long long)N, C1);

  // ---- GEMM2: h2 = x1 @ W2  (50000x256x40, padded to 48) ----
  {
    long long tiles = ((N + 15) / 16) * (long long)NT2;
    gemm_bf16_wmma<<<gridFor(tiles, 8), TPB, 0, stream>>>(x1b, W2pk, h2, N, C1, NT2, NCLS, NCLS);
  }

  // ---- layer-2 attention (heads=1, C=40) ----
  att_dot<<<gridFor((long long)N, TPB), TPB, 0, stream>>>(h2, attS2, as2, N, 1, NCLS);
  att_dot<<<gridFor((long long)N, TPB), TPB, 0, stream>>>(h2, attD2, ad2, N, 1, NCLS);
  fill_f32<<<gridFor((long long)N, TPB), TPB, 0, stream>>>(m2, -INFINITY, (long long)N);
  fill_f32<<<gridFor((long long)N, TPB), TPB, 0, stream>>>(z2, 0.f, (long long)N);
  fill_f32<<<gridFor((long long)N * NCLS, TPB), TPB, 0, stream>>>(out2, 0.f, (long long)N * NCLS);
  edge_max   <<<gridFor(ETOT, TPB), TPB, 0, stream>>>(ei, as2, ad2, m2, 1, ETOT, E);
  edge_expsum<<<gridFor(ETOT, TPB), TPB, 0, stream>>>(ei, as2, ad2, m2, z2, 1, ETOT, E);
  edge_aggr4 <<<gridFor(ETOT * (NCLS / 4), TPB), TPB, 0, stream>>>(ei, as2, ad2, m2, z2, h2, out2,
                                                                   1, NCLS, ETOT, E);

  // ---- skip GEMM: hskip = x @ W_skip (50000x128x40, padded to 48) ----
  {
    long long tiles = ((N + 15) / 16) * (long long)NT2;
    gemm_bf16_wmma<<<gridFor(tiles, 8), TPB, 0, stream>>>(xb, Wskippk, hskip, N, KIN, NT2, NCLS, NCLS);
  }

  // ---- combine + log_softmax ----
  final_logsoftmax<<<gridFor((long long)N, TPB), TPB, 0, stream>>>(out2, hskip, b2, bskip, out, N);
}
